// IPA_37409165148291
// MI455X (gfx1250) — matrix operand order repaired and checked by reading, not compiled, not run on hardware
//
#include <hip/hip_runtime.h>
#include <hip/hip_bf16.h>

typedef __attribute__((ext_vector_type(16))) _Float16 v16h;
typedef __attribute__((ext_vector_type(8)))  float    v8f;

// Sizes fixed by the reference
#define NN   768      // nodes per graph
#define NE   12288    // edges per graph
#define HID  64
#define NBAT 8

// ---------------------------------------------------------------------------
// activations
// mila(x) = x*tanh(softplus(x-1)).  tanh(log w) = (w^2-1)/(w^2+1), w=1+e^(x-1)
__device__ __forceinline__ float milaf(float x) {
  float t  = __expf(fminf(x - 1.0f, 30.0f));   // clamp avoids inf/inf
  float w  = 1.0f + t;
  float w2 = w * w;
  return x * (w2 - 1.0f) * __builtin_amdgcn_rcpf(w2 + 1.0f);
}
__device__ __forceinline__ float softplusf(float z) {
  float e  = __expf(fminf(z, 20.0f));
  float sp = __logf(1.0f + e);
  return (z > 20.0f) ? z : sp;
}

// ---------------------------------------------------------------------------
__global__ void k_zero(float* __restrict__ p, int n) {
  int i = blockIdx.x * blockDim.x + threadIdx.x;
  if (i < n) p[i] = 0.0f;
}

// single-layer embedding MLP: out[row,h] = mila(in[row,:]@W + b), block=64
__global__ void k_emb(const float* __restrict__ in, const float* __restrict__ W,
                      const float* __restrict__ b, float* __restrict__ out, int K) {
  int row = blockIdx.x, h = threadIdx.x;
  float acc = b[h];
  for (int k = 0; k < K; ++k) acc += in[row * K + k] * W[k * 64 + h];
  out[row * 64 + h] = milaf(acc);
}

// fused edge block: x=[edge(64)|node[recv](64)|node[send](64)] -> 64 -> 64
__global__ __launch_bounds__(256) void k_edge_block(
    const float* __restrict__ edges, const float* __restrict__ nodes,
    const int* __restrict__ recv, const int* __restrict__ send,
    const float* __restrict__ W1, const float* __restrict__ b1,
    const float* __restrict__ W2, const float* __restrict__ b2,
    float* __restrict__ out) {
  __shared__ float x[4][192];
  __shared__ float y1[4][64];
  int tid = threadIdx.x;
  int el = tid >> 6, h = tid & 63;
  int e = blockIdx.x * 4 + el;
  int rn = recv[e], sn = send[e];
  x[el][h]       = edges[e * 64 + h];
  x[el][64 + h]  = nodes[rn * 64 + h];
  x[el][128 + h] = nodes[sn * 64 + h];
  __syncthreads();
  float acc = b1[h];
#pragma unroll 8
  for (int k = 0; k < 192; ++k) acc += x[el][k] * W1[k * 64 + h];
  y1[el][h] = milaf(acc);
  __syncthreads();
  float acc2 = b2[h];
#pragma unroll 8
  for (int k = 0; k < 64; ++k) acc2 += y1[el][k] * W2[k * 64 + h];
  out[e * 64 + h] = milaf(acc2);
}

// segment_sum via float atomics: agg[recv[e],h] += edges[e,h]
__global__ void k_scatter(const float* __restrict__ edges,
                          const int* __restrict__ recv,
                          float* __restrict__ agg) {
  int idx = blockIdx.x * blockDim.x + threadIdx.x;  // NE*64 threads
  int e = idx >> 6, h = idx & 63;
  atomicAdd(&agg[recv[e] * 64 + h], edges[idx]);
}

// fused node block: x=[agg(64)|node(64)] -> 64 -> 64
__global__ __launch_bounds__(256) void k_node_block(
    const float* __restrict__ agg, const float* __restrict__ nodes,
    const float* __restrict__ W1, const float* __restrict__ b1,
    const float* __restrict__ W2, const float* __restrict__ b2,
    float* __restrict__ out) {
  __shared__ float x[4][128];
  __shared__ float y1[4][64];
  int tid = threadIdx.x;
  int nl = tid >> 6, h = tid & 63;
  int n = blockIdx.x * 4 + nl;
  x[nl][h]      = agg[n * 64 + h];
  x[nl][64 + h] = nodes[n * 64 + h];
  __syncthreads();
  float acc = b1[h];
#pragma unroll 8
  for (int k = 0; k < 128; ++k) acc += x[nl][k] * W1[k * 64 + h];
  y1[nl][h] = milaf(acc);
  __syncthreads();
  float acc2 = b2[h];
#pragma unroll 8
  for (int k = 0; k < 64; ++k) acc2 += y1[nl][k] * W2[k * 64 + h];
  out[n * 64 + h] = milaf(acc2);
}

// pp first layer factorization: U = nf@W0_top + b0/2, V = nf@W0_bot + b0/2
// which: 0->U1(n1,top) 1->V1(n1,bot) 2->U2(n2,top) 3->V2(n2,bot)
__global__ void k_uv(const float* __restrict__ n1f, const float* __restrict__ n2f,
                     const float* __restrict__ W0, const float* __restrict__ b0,
                     float* __restrict__ U1, float* __restrict__ V1,
                     float* __restrict__ U2, float* __restrict__ V2) {
  int row = blockIdx.x, h = threadIdx.x, which = blockIdx.y;
  const float* src = (which & 2) ? n2f : n1f;
  const float* W = W0 + ((which & 1) ? 64 * 64 : 0);
  float acc = 0.5f * b0[h];
  for (int k = 0; k < 64; ++k) acc += src[row * 64 + k] * W[k * 64 + h];
  float* dst = (which == 0) ? U1 : (which == 1) ? V1 : (which == 2) ? U2 : V2;
  dst[row * 64 + h] = acc;
}

// Pack W1 (pp hidden, 64x64 row-major, K x N) into the 8 WMMA B fragments
// (kt in 0..1, nt in 0..3).  B packed in contiguous-k order (A matches): lane L, half h:
//   K = kt*32 + (L<16?0:16) + h,  N = nt*16 + (L&15)
__global__ void k_packw2(const float* __restrict__ W2, _Float16* __restrict__ frag) {
  int tid = threadIdx.x;  // 256 = 8 frags * 32 lanes
  int f = tid >> 5, lane = tid & 31;
  int kt = f >> 2, nt = f & 3;
  int koff = kt * 32 + ((lane < 16) ? 0 : 16);
  int n = nt * 16 + (lane & 15);
  for (int h = 0; h < 16; ++h)
    frag[tid * 16 + h] = (_Float16)W2[(koff + h) * 64 + n];
}

// ---------------------------------------------------------------------------
// Pair kernel.  Block = 256 thr = 8 waves; wave w handles pairs
// (i0..i0+15) x (j = blockIdx.y*8 + w).  Hidden 64x64 GEMM per direction via
// 8x v_wmma_f32_16x16x32_f16 with W1 held in VGPRs as B fragments.
__global__ __launch_bounds__(256) void k_pair(
    const float* __restrict__ U1, const float* __restrict__ V2,
    const float* __restrict__ V1, const float* __restrict__ U2,
    const _Float16* __restrict__ w2frag,
    const float* __restrict__ b2, const float* __restrict__ W3,
    const float* __restrict__ b3, const float* __restrict__ dist,
    float* __restrict__ partial) {
  __shared__ float sAcc[16];           // ex[8] | disp[8] per workgroup
  __shared__ float sPred[8][16][2];    // per-wave (C6,C9) for 16 pairs
  int tid = threadIdx.x;
  if (tid < 16) sAcc[tid] = 0.0f;
  __syncthreads();

  int wave = tid >> 5, lane = tid & 31;
  int i0 = blockIdx.x * 16;
  int j  = blockIdx.y * 8 + wave;
  int mrow  = lane & 15;               // A-fragment row for this lane
  int kbase = (lane < 16) ? 0 : 16;    // A-fragment K offset for upper half
  int ncol  = lane & 15;               // C-fragment column for this lane

  // W1 B-fragments: 8 x v16h, resident in VGPRs for the whole kernel
  v16h Bf[8];
#pragma unroll
  for (int f = 0; f < 8; ++f)
    Bf[f] = *(const v16h*)(w2frag + (f * 32 + lane) * 16);

  float cb[4], w30[4], w31[4];
#pragma unroll
  for (int nt = 0; nt < 4; ++nt) {
    int n = nt * 16 + ncol;
    cb[nt]  = b2[n];                   // bias folded into C init
    w30[nt] = W3[n * 2 + 0];
    w31[nt] = W3[n * 2 + 1];
  }
  float b30 = b3[0], b31 = b3[1];

  float predC6[8], predC9[8];
#pragma unroll
  for (int v = 0; v < 8; ++v) { predC6[v] = 0.0f; predC9[v] = 0.0f; }

#pragma unroll
  for (int dir = 0; dir < 2; ++dir) {
    // dir0: [f1|f2]@W0 = U1[i]+V2[j];  dir1: [f2|f1]@W0 = U2[j]+V1[i]
    const float* Urow = ((dir == 0) ? U1 : V1) + (size_t)(i0 + mrow) * 64;
    const float* Vrow = ((dir == 0) ? V2 : U2) + (size_t)j * 64;

    // build two A fragments (K=0..31, K=32..63), mila applied per pair
    v16h af[2];
#pragma unroll
    for (int f = 0; f < 2; ++f) {
#pragma unroll
      for (int g = 0; g < 2; ++g) {
        int k0 = f * 32 + kbase + g * 8;
        float4 u0 = *(const float4*)(Urow + k0);
        float4 u1 = *(const float4*)(Urow + k0 + 4);
        float4 v0 = *(const float4*)(Vrow + k0);
        float4 v1 = *(const float4*)(Vrow + k0 + 4);
        af[f][g * 8 + 0] = (_Float16)milaf(u0.x + v0.x);
        af[f][g * 8 + 1] = (_Float16)milaf(u0.y + v0.y);
        af[f][g * 8 + 2] = (_Float16)milaf(u0.z + v0.z);
        af[f][g * 8 + 3] = (_Float16)milaf(u0.w + v0.w);
        af[f][g * 8 + 4] = (_Float16)milaf(u1.x + v1.x);
        af[f][g * 8 + 5] = (_Float16)milaf(u1.y + v1.y);
        af[f][g * 8 + 6] = (_Float16)milaf(u1.z + v1.z);
        af[f][g * 8 + 7] = (_Float16)milaf(u1.w + v1.w);
      }
    }

    float t0[8], t1[8];
#pragma unroll
    for (int v = 0; v < 8; ++v) { t0[v] = 0.0f; t1[v] = 0.0f; }

#pragma unroll
    for (int nt = 0; nt < 4; ++nt) {
      v8f c;
#pragma unroll
      for (int v = 0; v < 8; ++v) c[v] = cb[nt];
      c = __builtin_amdgcn_wmma_f32_16x16x32_f16(false, af[0], false, Bf[0 + nt],
                                                 (short)0, c, false, false);
      c = __builtin_amdgcn_wmma_f32_16x16x32_f16(false, af[1], false, Bf[4 + nt],
                                                 (short)0, c, false, false);
      // epilogue: mila + partial 64->2 dot (this lane's 4 columns)
#pragma unroll
      for (int v = 0; v < 8; ++v) {
        float hh = milaf(c[v]);
        t0[v] += hh * w30[nt];
        t1[v] += hh * w31[nt];
      }
    }
    // reduce over the 16 lanes sharing an m-half (xor<=8 stays in-half)
#pragma unroll
    for (int off = 1; off <= 8; off <<= 1) {
#pragma unroll
      for (int v = 0; v < 8; ++v) {
        t0[v] += __shfl_xor(t0[v], off, 32);
        t1[v] += __shfl_xor(t1[v], off, 32);
      }
    }
#pragma unroll
    for (int v = 0; v < 8; ++v) {
      predC6[v] += softplusf(t0[v] + b30);
      predC9[v] += softplusf(t1[v] + b31);
    }
  }

  // publish preds: lane0 holds m=0..7 (C VGPR v -> m=v), lane16 holds m=8..15
  if (lane == 0 || lane == 16) {
    int mb = (lane == 0) ? 0 : 8;
#pragma unroll
    for (int v = 0; v < 8; ++v) {
      sPred[wave][mb + v][0] = predC6[v];
      sPred[wave][mb + v][1] = predC9[v];
    }
  }
  __syncthreads();

  // dispersion sums: 16 m x 8 batches = 128 tasks spread over 32 lanes
#pragma unroll
  for (int q = 0; q < 4; ++q) {
    int t  = lane + q * 32;
    int mm = t >> 3;
    int bb = t & 7;
    float C6 = sPred[wave][mm][0];
    float C9 = sPred[wave][mm][1];
    int i = i0 + mm;
    float r   = dist[((size_t)bb * NN + i) * NN + j];
    float inv = __builtin_amdgcn_rcpf(r);
    float i3 = inv * inv * inv;
    float i6 = i3 * i3;
    float i9 = i6 * i3;
    bool  ok  = (r < 10.0f);
    float exv = ok ?  C9 * i9 : 0.0f;
    float dv  = ok ? -C6 * i6 : 0.0f;
    atomicAdd(&sAcc[bb], exv);
    atomicAdd(&sAcc[8 + bb], dv);
  }
  __syncthreads();
  if (tid < 16)
    partial[(size_t)(blockIdx.x * gridDim.y + blockIdx.y) * 16 + tid] = sAcc[tid];
}

// deterministic final reduction: partial[nWG][16] -> out[24] = (ex+disp, ex, disp)
__global__ void k_reduce(const float* __restrict__ partial, float* __restrict__ out,
                         int nWG) {
  __shared__ float s[16][16];
  __shared__ float fin[16];
  int tid = threadIdx.x;  // 256
  int col = tid & 15, seg = tid >> 4;
  float a = 0.0f;
  for (int r = seg; r < nWG; r += 16) a += partial[(size_t)r * 16 + col];
  s[seg][col] = a;
  __syncthreads();
  if (tid < 16) {
    float t = 0.0f;
    for (int k = 0; k < 16; ++k) t += s[k][tid];
    fin[tid] = t;
  }
  __syncthreads();
  if (tid < 8) {
    float ex = fin[tid], dp = fin[8 + tid];
    out[tid]      = ex + dp;
    out[8 + tid]  = ex;
    out[16 + tid] = dp;
  }
}

// ---------------------------------------------------------------------------
extern "C" void kernel_launch(void* const* d_in, const int* in_sizes, int n_in,
                              void* d_out, int out_size, void* d_ws, size_t ws_size,
                              hipStream_t stream) {
  (void)in_sizes; (void)n_in; (void)out_size; (void)ws_size;
  auto F = [&](int i) { return (const float*)d_in[i]; };
  // setup_inputs() dict insertion order, params flattened in insertion order:
  const float* nodes_1 = F(0);
  const float* edges_1 = F(1);
  const float* nodes_2 = F(2);
  const float* edges_2 = F(3);
  const float* dist    = F(4);
  const float* embnW = F(5); const float* embnB = F(6);   // emb_node
  const float* embeW = F(7); const float* embeB = F(8);   // emb_edge
  // gn_edge: idx 9 + 4*s {W1,b1,W2,b2}; gn_node: idx 21 + 4*s
  const float* ppW0 = F(33); const float* ppB0 = F(34);
  const float* ppW1 = F(35); const float* ppB1 = F(36);
  const float* ppW2 = F(37); const float* ppB2 = F(38);
  const int* send1 = (const int*)d_in[39];
  const int* recv1 = (const int*)d_in[40];
  const int* send2 = (const int*)d_in[41];
  const int* recv2 = (const int*)d_in[42];

  // workspace layout (floats) — ~14.7 MB total
  float* ws  = (float*)d_ws;
  float* n1a = ws;                 float* n1b = n1a + NN * 64;
  float* n2a = n1b + NN * 64;      float* n2b = n2a + NN * 64;
  float* e1a = n2b + NN * 64;      float* e1b = e1a + NE * 64;
  float* e2a = e1b + NE * 64;      float* e2b = e2a + NE * 64;
  float* agg = e2b + NE * 64;
  float* U1 = agg + NN * 64;  float* V1 = U1 + NN * 64;
  float* U2 = V1 + NN * 64;   float* V2 = U2 + NN * 64;
  float* partial = V2 + NN * 64;            // 4608 * 16
  _Float16* w2frag = (_Float16*)(partial + 4608 * 16);

  // ---- graph phase (both graphs, 3 interaction steps each) ----
  for (int g = 0; g < 2; ++g) {
    const float* nin = g ? nodes_2 : nodes_1;
    const float* ein = g ? edges_2 : edges_1;
    const int* snd = g ? send2 : send1;
    const int* rcv = g ? recv2 : recv1;
    float* na = g ? n2a : n1a;  float* nb = g ? n2b : n1b;
    float* ea = g ? e2a : e1a;  float* eb = g ? e2b : e1b;
    k_emb<<<NN, 64, 0, stream>>>(nin, embnW, embnB, na, 16);
    k_emb<<<NE, 64, 0, stream>>>(ein, embeW, embeB, ea, 8);
    for (int s = 0; s < 3; ++s) {
      const float* We1 = F(9 + 4 * s);  const float* be1 = F(10 + 4 * s);
      const float* We2 = F(11 + 4 * s); const float* be2 = F(12 + 4 * s);
      const float* Wn1 = F(21 + 4 * s); const float* bn1 = F(22 + 4 * s);
      const float* Wn2 = F(23 + 4 * s); const float* bn2 = F(24 + 4 * s);
      k_edge_block<<<NE / 4, 256, 0, stream>>>(ea, na, rcv, snd, We1, be1, We2, be2, eb);
      k_zero<<<(NN * 64 + 255) / 256, 256, 0, stream>>>(agg, NN * 64);
      k_scatter<<<(NE * 64) / 256, 256, 0, stream>>>(eb, rcv, agg);
      k_node_block<<<NN / 4, 256, 0, stream>>>(agg, na, Wn1, bn1, Wn2, bn2, nb);
      float* t = na; na = nb; nb = t;
      t = ea; ea = eb; eb = t;
    }
  }
  // after 3 steps (odd #swaps) final node features live in n1b / n2b
  k_uv<<<dim3(NN, 4), 64, 0, stream>>>(n1b, n2b, ppW0, ppB0, U1, V1, U2, V2);
  k_packw2<<<1, 256, 0, stream>>>(ppW1, w2frag);

  // ---- pair phase: 48 x 96 workgroups, 8 waves each, 16 pairs/wave ----
  k_pair<<<dim3(48, 96), 256, 0, stream>>>(U1, V2, V1, U2, w2frag,
                                           ppB1, ppW2, ppB2, dist, partial);
  k_reduce<<<1, 256, 0, stream>>>(partial, (float*)d_out, 48 * 96);
}
